// GroupQuerySelfAttention_14955076124980
// MI455X (gfx1250) — matrix operand order repaired and checked by reading, not compiled, not run on hardware
//
#include <hip/hip_runtime.h>
#include <stdint.h>

typedef __attribute__((ext_vector_type(16))) __bf16 v16bf;
typedef __attribute__((ext_vector_type(8)))  float  v8f;

#define B_   2
#define S_   2048
#define D_   768
#define NH_  12
#define GH_  4
#define HD_  64
#define GD_  256
#define REP_ 3
#define K_   768          // contraction dim of all projection GEMMs (constant -> imm offsets)
#define SCALE_ 0.125f
#define MASKF_ (-1e-9f)

__device__ __forceinline__ uint16_t f2bf(float f) {
    uint32_t u = __float_as_uint(f);
    u += 0x7fffu + ((u >> 16) & 1u);   // round-to-nearest-even
    return (uint16_t)(u >> 16);
}
__device__ __forceinline__ uint32_t pk2(float a, float b) {
    return (uint32_t)f2bf(a) | ((uint32_t)f2bf(b) << 16);
}

union FragAB { v16bf v; uint32_t u[8]; uint4 q[2]; };
union FragC  { v8f  v; float    f[8]; };

__device__ __forceinline__ uint4 ld128(const uint16_t* p) { return *(const uint4*)p; }

// ---------------- conversion kernels ----------------
__global__ void conv_bf16(const float* __restrict__ in, uint16_t* __restrict__ out, int n) {
    int i = blockIdx.x * blockDim.x + threadIdx.x;
    if (i < n) out[i] = f2bf(in[i]);
}

// in: fp32 [K][N] row-major -> out: bf16 [N][K] (transposed)
__global__ void conv_T(const float* __restrict__ in, uint16_t* __restrict__ out, int K, int N) {
    int i = blockIdx.x * blockDim.x + threadIdx.x;
    if (i < K * N) {
        int k = i / N, n = i % N;
        out[(size_t)n * K + k] = f2bf(in[i]);
    }
}

// ---------------- WMMA GEMM: C[M,N] = A[M,768] @ W[768,N] + bias ----------------
// One wave computes a 16x128 tile (A-fragment reused across 8 WMMAs per K-step).
// A: bf16 [M][768] row-major.  WT: bf16 [N][768] (W transposed).
// MODE 0: bf16 row-major [M][N] | MODE 1: fp32 row-major | MODE 2: V^T scatter (b128 packed)
template <int MODE>
__global__ __launch_bounds__(32, 1)
void gemm_wmma(const uint16_t* __restrict__ A, const uint16_t* __restrict__ WT,
               const float* __restrict__ bias, void* __restrict__ outp, int N) {
    const int lane = threadIdx.x & 31;
    const int hf   = lane >> 4;
    const int ln   = lane & 15;
    const int mbase = blockIdx.x * 16;
    const int nbase = blockIdx.y * 128;

    FragC acc[8];
#pragma unroll
    for (int c = 0; c < 8; ++c)
#pragma unroll
        for (int r = 0; r < 8; ++r) acc[c].f[r] = 0.0f;

    // per-lane base pointers; all further offsets are compile-time immediates
    const uint16_t* ap = A  + (size_t)(mbase + ln) * K_ + 8 * hf;    // A-frag runs
    const uint16_t* wp = WT + (size_t)(nbase + ln) * K_ + 8 * hf;    // B-frag runs

    for (int k0 = 0; k0 < K_; k0 += 32) {
        FragAB a;
        a.q[0] = ld128(ap);        // elements 8hf .. 8hf+7
        a.q[1] = ld128(ap + 16);   // elements 16+8hf .. 16+8hf+7
        ap += 32;
#pragma unroll
        for (int c = 0; c < 8; ++c) {
            FragAB bfr;
            bfr.q[0] = ld128(wp + c * 16 * K_);       // elements 8hf .. 8hf+7
            bfr.q[1] = ld128(wp + c * 16 * K_ + 16);  // elements 16+8hf .. 16+8hf+7
            acc[c].v = __builtin_amdgcn_wmma_f32_16x16x32_bf16(
                false, a.v, false, bfr.v, (short)0, acc[c].v, false, false);
        }
        wp += 32;
    }

#pragma unroll
    for (int c = 0; c < 8; ++c) {
        const int col = nbase + c * 16 + ln;
        const float bv = bias[col];
        if (MODE == 2) {
            // 8 consecutive s-values per lane -> one packed b128 store
            const int row0 = mbase + 8 * hf;               // rows row0..row0+7
            const int b = row0 >> 11, s0 = row0 & (S_ - 1);
            const int g = col >> 6,  d  = col & (HD_ - 1);
            uint4 w;
            w.x = pk2(acc[c].f[0] + bv, acc[c].f[1] + bv);
            w.y = pk2(acc[c].f[2] + bv, acc[c].f[3] + bv);
            w.z = pk2(acc[c].f[4] + bv, acc[c].f[5] + bv);
            w.w = pk2(acc[c].f[6] + bv, acc[c].f[7] + bv);
            uint16_t* dst = (uint16_t*)outp + (((size_t)(b * GH_ + g) * HD_ + d) << 11) + s0;
            *(uint4*)dst = w;
        } else {
#pragma unroll
            for (int r = 0; r < 8; ++r) {
                const int row = mbase + r + 8 * hf;
                const float val = acc[c].f[r] + bv;
                if (MODE == 0) ((uint16_t*)outp)[(size_t)row * N + col] = f2bf(val);
                else           ((float*)outp)[(size_t)row * N + col]    = val;
            }
        }
    }
}

// ---------------- fused flash attention: one wave = 32 queries of one head ----------------
// K/V fragments are loaded once and reused across both 16-query tiles.
// Qb: bf16 [B*S][D]; Kb: bf16 [B*S][GD]; VTb: bf16 [(b*GH+g)*HD+d][S]
// merged: bf16 in the reference's buggy-merge layout [(b*D + h*HD + d)][S]
__global__ __launch_bounds__(32, 1)
void attn_wmma(const uint16_t* __restrict__ Qb, const uint16_t* __restrict__ Kb,
               const uint16_t* __restrict__ VTb, uint16_t* __restrict__ merged) {
    __shared__ uint16_t Plds[2 * 16 * 32];
    const int lane = threadIdx.x & 31;
    const int hf   = lane >> 4;
    const int ln   = lane & 15;
    const int qt = blockIdx.x, h = blockIdx.y, b = blockIdx.z;
    const int g = h / REP_;
    const int qbase0 = qt * 32;

    // Q tiles (2 x 16x64) as A-fragments; two b128 loads per fragment
    FragAB qa[2][2];
#pragma unroll
    for (int mt = 0; mt < 2; ++mt) {
        const uint16_t* qp =
            Qb + ((size_t)b * S_ + qbase0 + mt * 16 + ln) * D_ + h * HD_ + 8 * hf;
#pragma unroll
        for (int f = 0; f < 2; ++f) {
            qa[mt][f].q[0] = ld128(qp + f * 32);
            qa[mt][f].q[1] = ld128(qp + f * 32 + 16);
        }
    }

    const uint16_t* kp = Kb  + ((size_t)b * S_ + ln) * GD_ + g * HD_ + 8 * hf;
    const uint16_t* vp = VTb + ((size_t)(b * GH_ + g) * HD_ + ln) * S_ + 8 * hf;

    float mrow[2][8], lrow[2][8];
    FragC o[2][4];
#pragma unroll
    for (int mt = 0; mt < 2; ++mt)
#pragma unroll
        for (int r = 0; r < 8; ++r) { mrow[mt][r] = -3.0e38f; lrow[mt][r] = 0.0f; }
#pragma unroll
    for (int mt = 0; mt < 2; ++mt)
#pragma unroll
        for (int c = 0; c < 4; ++c)
#pragma unroll
            for (int r = 0; r < 8; ++r) o[mt][c].f[r] = 0.0f;

    for (int kt = 0; kt < S_ / 32; ++kt) {
        const int kbase = kt * 32;
        FragC s[2][2];
#pragma unroll
        for (int mt = 0; mt < 2; ++mt)
#pragma unroll
            for (int sub = 0; sub < 2; ++sub)
#pragma unroll
                for (int r = 0; r < 8; ++r) s[mt][sub].f[r] = 0.0f;

        // scores: 32 queries x 32 keys; K-fragments shared by both query tiles
#pragma unroll
        for (int sub = 0; sub < 2; ++sub)
#pragma unroll
            for (int f = 0; f < 2; ++f) {
                FragAB bk;
                bk.q[0] = ld128(kp + sub * 16 * GD_ + f * 32);
                bk.q[1] = ld128(kp + sub * 16 * GD_ + f * 32 + 16);
#pragma unroll
                for (int mt = 0; mt < 2; ++mt)
                    s[mt][sub].v = __builtin_amdgcn_wmma_f32_16x16x32_bf16(
                        false, qa[mt][f].v, false, bk.v, (short)0, s[mt][sub].v, false, false);
            }
        kp += 32 * GD_;

        // scale + (buggy) mask fill: masked logits are exactly -1e-9, still inside softmax
#pragma unroll
        for (int mt = 0; mt < 2; ++mt)
#pragma unroll
            for (int sub = 0; sub < 2; ++sub) {
                const int keyg = kbase + sub * 16 + ln;
#pragma unroll
                for (int r = 0; r < 8; ++r) {
                    const int qg = qbase0 + mt * 16 + r + 8 * hf;
                    s[mt][sub].f[r] = (keyg <= qg) ? s[mt][sub].f[r] * SCALE_ : MASKF_;
                }
            }

        __syncthreads();
        // online softmax per row; rows reduce across the 16 lanes of each half
#pragma unroll
        for (int mt = 0; mt < 2; ++mt)
#pragma unroll
            for (int r = 0; r < 8; ++r) {
                float tm = fmaxf(s[mt][0].f[r], s[mt][1].f[r]);
#pragma unroll
                for (int off = 8; off >= 1; off >>= 1) tm = fmaxf(tm, __shfl_xor(tm, off, 32));
                const float mnew  = fmaxf(mrow[mt][r], tm);
                const float alpha = __expf(mrow[mt][r] - mnew);
                const float p0 = __expf(s[mt][0].f[r] - mnew);
                const float p1 = __expf(s[mt][1].f[r] - mnew);
                float rs = p0 + p1;
#pragma unroll
                for (int off = 8; off >= 1; off >>= 1) rs += __shfl_xor(rs, off, 32);
                lrow[mt][r] = lrow[mt][r] * alpha + rs;
                mrow[mt][r] = mnew;
#pragma unroll
                for (int c = 0; c < 4; ++c) o[mt][c].f[r] *= alpha;
                const int qr = r + 8 * hf;
                Plds[mt * 512 + qr * 32 + ln]      = f2bf(p0);
                Plds[mt * 512 + qr * 32 + 16 + ln] = f2bf(p1);
            }
        __syncthreads();

        // P tiles back as A-fragments via LDS (2 ds_load_b128 each)
        FragAB pa[2];
#pragma unroll
        for (int mt = 0; mt < 2; ++mt) {
            const uint16_t* pp = Plds + mt * 512 + ln * 32 + 8 * hf;
            pa[mt].q[0] = ld128(pp);
            pa[mt].q[1] = ld128(pp + 16);
        }
        // O += P @ V ; V fragments shared by both query tiles
#pragma unroll
        for (int c = 0; c < 4; ++c) {
            FragAB bv;
            bv.q[0] = ld128(vp + c * 16 * S_);
            bv.q[1] = ld128(vp + c * 16 * S_ + 16);
#pragma unroll
            for (int mt = 0; mt < 2; ++mt)
                o[mt][c].v = __builtin_amdgcn_wmma_f32_16x16x32_bf16(
                    false, pa[mt].v, false, bv.v, (short)0, o[mt][c].v, false, false);
        }
        vp += 32;
        __syncthreads();
    }

    // normalize + packed b128 stores into buggy-merge layout [b][(h*64+d)][q]
#pragma unroll
    for (int mt = 0; mt < 2; ++mt) {
        float rinv[8];
#pragma unroll
        for (int r = 0; r < 8; ++r) rinv[r] = 1.0f / lrow[mt][r];
#pragma unroll
        for (int c = 0; c < 4; ++c) {
            const int d = c * 16 + ln;
            uint4 w;
            w.x = pk2(o[mt][c].f[0] * rinv[0], o[mt][c].f[1] * rinv[1]);
            w.y = pk2(o[mt][c].f[2] * rinv[2], o[mt][c].f[3] * rinv[3]);
            w.z = pk2(o[mt][c].f[4] * rinv[4], o[mt][c].f[5] * rinv[5]);
            w.w = pk2(o[mt][c].f[6] * rinv[6], o[mt][c].f[7] * rinv[7]);
            uint16_t* dst = merged + ((size_t)b * D_ + h * HD_ + d) * S_
                          + qbase0 + mt * 16 + 8 * hf;
            *(uint4*)dst = w;
        }
    }
}

// ---------------- launch ----------------
extern "C" void kernel_launch(void* const* d_in, const int* in_sizes, int n_in,
                              void* d_out, int out_size, void* d_ws, size_t ws_size,
                              hipStream_t stream) {
    const float* x  = (const float*)d_in[0];
    // d_in[1] = masks (causal tril) — computed analytically, not loaded
    const float* Wq = (const float*)d_in[2];
    const float* bq = (const float*)d_in[3];
    const float* Wk = (const float*)d_in[4];
    const float* bk = (const float*)d_in[5];
    const float* Wv = (const float*)d_in[6];
    const float* bv = (const float*)d_in[7];
    const float* Wo = (const float*)d_in[8];
    const float* bo = (const float*)d_in[9];

    uint16_t* ws = (uint16_t*)d_ws;
    size_t off = 0;
    uint16_t* xb  = ws + off; off += (size_t)B_ * S_ * D_;   // x bf16
    uint16_t* WqT = ws + off; off += (size_t)D_ * D_;        // W^T bf16 [N][K]
    uint16_t* WkT = ws + off; off += (size_t)D_ * GD_;
    uint16_t* WvT = ws + off; off += (size_t)D_ * GD_;
    uint16_t* WoT = ws + off; off += (size_t)D_ * D_;
    uint16_t* Qb  = ws + off; off += (size_t)B_ * S_ * D_;   // Q bf16 [B*S][D]
    uint16_t* Kb  = ws + off; off += (size_t)B_ * S_ * GD_;  // K bf16 [B*S][GD]
    uint16_t* VTb = ws + off; off += (size_t)B_ * S_ * GD_;  // V^T bf16
    uint16_t* Mg  = ws + off; off += (size_t)B_ * S_ * D_;   // merged bf16

    {
        int n = B_ * S_ * D_;
        conv_bf16<<<(n + 255) / 256, 256, 0, stream>>>(x, xb, n);
    }
    conv_T<<<(D_ * D_  + 255) / 256, 256, 0, stream>>>(Wq, WqT, D_, D_);
    conv_T<<<(D_ * GD_ + 255) / 256, 256, 0, stream>>>(Wk, WkT, D_, GD_);
    conv_T<<<(D_ * GD_ + 255) / 256, 256, 0, stream>>>(Wv, WvT, D_, GD_);
    conv_T<<<(D_ * D_  + 255) / 256, 256, 0, stream>>>(Wo, WoT, D_, D_);

    dim3 gq(B_ * S_ / 16, D_ / 128);   // 256 x 6
    dim3 gk(B_ * S_ / 16, GD_ / 128);  // 256 x 2
    gemm_wmma<0><<<gq, 32, 0, stream>>>(xb, WqT, bq, Qb,  D_);
    gemm_wmma<0><<<gk, 32, 0, stream>>>(xb, WkT, bk, Kb,  GD_);
    gemm_wmma<2><<<gk, 32, 0, stream>>>(xb, WvT, bv, VTb, GD_);

    dim3 ga(S_ / 32, NH_, B_);         // 64 x 12 x 2
    attn_wmma<<<ga, 32, 0, stream>>>(Qb, Kb, VTb, Mg);

    gemm_wmma<1><<<gq, 32, 0, stream>>>(Mg, WoT, bo, d_out, D_);
}
